// HaarSymHQLinear_40252433498284
// MI455X (gfx1250) — matrix operand order repaired and checked by reading, not compiled, not run on hardware
//
#include <hip/hip_runtime.h>
#include <hip/hip_bf16.h>

// ---------------------------------------------------------------------------
// Problem dims
// ---------------------------------------------------------------------------
#define OUT_DIM   4096
#define IN_DIM    4096
#define M_TOTAL   8192            // 4 * 2048 rows of x
#define INV_SQRT2 0.70710678118654752440f

// GEMM tiling
#define MT 128                     // block tile M
#define NT 128                     // block tile N
#define KT 32                      // K step (bf16 WMMA K)
#define SKPAD 8                    // LDS row padding (bf16 elems) -> stride 40
#define SK (KT + SKPAD)

typedef __attribute__((ext_vector_type(16))) __bf16 v16bf;
typedef __attribute__((ext_vector_type(8)))  float  v8f;
typedef __attribute__((ext_vector_type(4)))  int    v4i;

// ---------------------------------------------------------------------------
// gfx1250 async global->LDS path (guarded; falls back to load + ds_store)
// Signature (probe-confirmed): (AS(1) v4i*, AS(3) v4i*, imm offset, imm cpol)
// ---------------------------------------------------------------------------
#if defined(__has_builtin)
#if __has_builtin(__builtin_amdgcn_global_load_async_to_lds_b128) && \
    __has_builtin(__builtin_amdgcn_s_wait_asynccnt)
#define USE_ASYNC_LDS 1
#endif
#endif
#ifndef USE_ASYNC_LDS
#define USE_ASYNC_LDS 0
#endif

__device__ __forceinline__ void copy16B_to_lds(const void* g, void* l) {
#if USE_ASYNC_LDS
  __builtin_amdgcn_global_load_async_to_lds_b128(
      (__attribute__((address_space(1))) v4i*)(void*)g,
      (__attribute__((address_space(3))) v4i*)(void*)l,
      /*offset=*/0, /*cpol=*/0);
#else
  *(v4i*)l = *(const v4i*)g;
#endif
}

// Wait until at most N async ops remain outstanding (in-order completion).
#if USE_ASYNC_LDS
#define ASYNC_WAIT(n) __builtin_amdgcn_s_wait_asynccnt(n)
#else
#define ASYNC_WAIT(n) ((void)0)
#endif

// ---------------------------------------------------------------------------
// Kernel 1: x fp32 -> bf16 (8 elems / thread)
// ---------------------------------------------------------------------------
__global__ __launch_bounds__(256) void cvt_x_kernel(const float* __restrict__ x,
                                                    __hip_bfloat16* __restrict__ xb_) {
  __bf16* xb = (__bf16*)xb_;
  const size_t base = ((size_t)blockIdx.x * 256 + threadIdx.x) * 8;
  const float4 f0 = ((const float4*)(x + base))[0];
  const float4 f1 = ((const float4*)(x + base))[1];
  __bf16 o[8] = {(__bf16)f0.x, (__bf16)f0.y, (__bf16)f0.z, (__bf16)f0.w,
                 (__bf16)f1.x, (__bf16)f1.y, (__bf16)f1.z, (__bf16)f1.w};
  *(v4i*)(xb + base) = *(const v4i*)o;
}

// ---------------------------------------------------------------------------
// Kernel 2: decode one weight row per block.
//   haar_freq[o,t] = codebook[indices[o*512 + t/8]][t%8] * (bit ? +1 : -1)
//   2-level inverse Haar (closed form):
//     k = i>>1, j = k>>1
//     a2 = (c[j] +- c[1024+j]);  w[i] = (a2*s +- c[2048+k]) * s * scale[o]
// ---------------------------------------------------------------------------
__global__ __launch_bounds__(512) void decode_w_kernel(
    const float* __restrict__ codebook, const float* __restrict__ scales,
    const int* __restrict__ indices, const int* __restrict__ signs_packed,
    __hip_bfloat16* __restrict__ w_) {
  __bf16* w = (__bf16*)w_;
  __shared__ float c[IN_DIM];
  const int o = blockIdx.x;
  const int t = threadIdx.x;
  const int g = o * (IN_DIM / 8) + t;
  const int idx = indices[g];
  const unsigned sp = (unsigned)signs_packed[g];
  const float4 cb0 = ((const float4*)codebook)[idx * 2 + 0];
  const float4 cb1 = ((const float4*)codebook)[idx * 2 + 1];
  const float vals[8] = {cb0.x, cb0.y, cb0.z, cb0.w, cb1.x, cb1.y, cb1.z, cb1.w};
#pragma unroll
  for (int e = 0; e < 8; ++e) {
    const float v = vals[e];
    c[t * 8 + e] = ((sp >> e) & 1u) ? v : -v;   // sign = 2*bit - 1
  }
  __syncthreads();
  const float sc = scales[o] * INV_SQRT2;
#pragma unroll
  for (int e = 0; e < 8; ++e) {
    const int i = t * 8 + e;
    const int k = i >> 1;
    const int j = k >> 1;
    const float a  = c[j];
    const float b  = c[1024 + j];
    const float d  = c[2048 + k];
    const float a2 = ((k & 1) ? (a - b) : (a + b)) * INV_SQRT2;
    const float v  = ((i & 1) ? (a2 - d) : (a2 + d)) * sc;
    w[(size_t)o * IN_DIM + i] = (__bf16)v;
  }
}

// ---------------------------------------------------------------------------
// Kernel 3: bf16 WMMA GEMM  out[m,n] = sum_k x[m,k] * w[n,k]
//   block: 256 threads = 8 waves; block tile 128(M) x 128(N)
//   wave (waveM 0..3, waveN 0..1): 32(M) x 64(N) = 2x4 WMMA accumulators
//   LDS double-buffered; async loads for tile i+1 overlap WMMA on tile i.
//   All fragments hoisted so the 12 ds_load_b128 clause precedes 8 WMMAs.
// ---------------------------------------------------------------------------
__global__ __launch_bounds__(256) void gemm_bf16_kernel(
    const __hip_bfloat16* __restrict__ xb_, const __hip_bfloat16* __restrict__ wb_,
    float* __restrict__ out) {
  const __bf16* xb = (const __bf16*)xb_;
  const __bf16* wb = (const __bf16*)wb_;

  __shared__ __align__(16) __bf16 xs[2 * MT * SK];   // 2 x 128 x 40 bf16 = 20 KB
  __shared__ __align__(16) __bf16 wt[2 * NT * SK];   // 2 x 128 x 40 bf16 = 20 KB

  const int tid   = threadIdx.x;
  const int lane  = tid & 31;
  const int wave  = tid >> 5;
  const int waveM = wave >> 1;          // 0..3 -> M offset waveM*32
  const int waveN = wave & 1;           // 0..1 -> N offset waveN*64
  const int gN0   = blockIdx.x * NT;
  const int gM0   = blockIdx.y * MT;

  // staging: both tiles are 128 rows x 32 k bf16; 16 elems (32B) per thread each
  const int sr   = tid >> 1;            // 0..127
  const int scol = (tid & 1) * 16;      // 0 or 16

  const __bf16* xsrc = xb + (size_t)(gM0 + sr) * IN_DIM + scol;
  const __bf16* wsrc = wb + (size_t)(gN0 + sr) * IN_DIM + scol;

  v8f acc[2][4] = {};

  // 4 async b128 loads per thread per stage
  auto stage = [&](int buf, int kk) {
    __bf16* xd = &xs[buf * (MT * SK) + sr * SK + scol];
    __bf16* wd = &wt[buf * (NT * SK) + sr * SK + scol];
    copy16B_to_lds(xsrc + kk,     xd);
    copy16B_to_lds(xsrc + kk + 8, xd + 8);
    copy16B_to_lds(wsrc + kk,     wd);
    copy16B_to_lds(wsrc + kk + 8, wd + 8);
  };

  // fragments in contiguous k order (same order for A and B):
  //  A (16x32): lane L -> row M=L&15; VGPRs hold K=16*(L>>4)+0..15
  //  B (32x16): lane L -> col N=L&15; VGPRs hold K=16*(L>>4)+0..15
  auto compute = [&](int buf) {
    const __bf16* xsb = &xs[buf * (MT * SK)];
    const __bf16* wtb = &wt[buf * (NT * SK)];
    // ---- load ALL fragments first (distinct registers -> one ds clause) ----
    v16bf afrag[2];
#pragma unroll
    for (int i = 0; i < 2; ++i) {
      const int m = waveM * 32 + i * 16 + (lane & 15);
      const __bf16* p = xsb + m * SK + ((lane >> 4) * 16);
#pragma unroll
      for (int t = 0; t < 8; ++t) {
        afrag[i][t]     = p[t];
        afrag[i][8 + t] = p[8 + t];
      }
    }
    v16bf bfrag[4];
#pragma unroll
    for (int j = 0; j < 4; ++j) {
      const int n = waveN * 64 + j * 16 + (lane & 15);
      const __bf16* p = wtb + n * SK + ((lane >> 4) * 16);
#pragma unroll
      for (int t = 0; t < 16; ++t) bfrag[j][t] = p[t];
    }
    // ---- 8 back-to-back WMMAs ----
#pragma unroll
    for (int j = 0; j < 4; ++j) {
      acc[0][j] = __builtin_amdgcn_wmma_f32_16x16x32_bf16(
          false, afrag[0], false, bfrag[j], (short)0, acc[0][j], false, false);
      acc[1][j] = __builtin_amdgcn_wmma_f32_16x16x32_bf16(
          false, afrag[1], false, bfrag[j], (short)0, acc[1][j], false, false);
    }
  };

  stage(0, 0);
  int buf = 0;
  for (int kk = 0; kk < IN_DIM - KT; kk += KT) {
    stage(buf ^ 1, kk + KT);                  // prefetch next tile into other buffer
    if (kk + 2 * KT < IN_DIM) {
      __builtin_prefetch(xsrc + kk + 2 * KT, 0, 1);   // global_prefetch_b8
      __builtin_prefetch(wsrc + kk + 2 * KT, 0, 1);
    }
    ASYNC_WAIT(4);                            // in-order: waits for *current* buffer only
    __syncthreads();
    compute(buf);
    __syncthreads();                          // protect buf from next iteration's stage
    buf ^= 1;
  }
  ASYNC_WAIT(0);                              // last tile
  __syncthreads();
  compute(buf);

  // ---- epilogue: C/D f32 layout: N = lane&15, M = r + 8*(lane>>4) ----
#pragma unroll
  for (int i = 0; i < 2; ++i) {
#pragma unroll
    for (int j = 0; j < 4; ++j) {
      const int n     = gN0 + waveN * 64 + j * 16 + (lane & 15);
      const int mbase = gM0 + waveM * 32 + i * 16 + ((lane >> 4) * 8);
#pragma unroll
      for (int r = 0; r < 8; ++r)
        out[(size_t)(mbase + r) * OUT_DIM + n] = acc[i][j][r];
    }
  }
}

// ---------------------------------------------------------------------------
// Host-side launch
// Inputs: 0=x(f32), 1=codebook(f32), 2=scales(f32), 3=indices(i32), 4=signs(i32)
// ws layout: [0, 64MiB) x_bf16 ; [64MiB, 96MiB) w_bf16
// ---------------------------------------------------------------------------
extern "C" void kernel_launch(void* const* d_in, const int* in_sizes, int n_in,
                              void* d_out, int out_size, void* d_ws, size_t ws_size,
                              hipStream_t stream) {
  (void)in_sizes; (void)n_in; (void)out_size; (void)ws_size;
  const float* x        = (const float*)d_in[0];
  const float* codebook = (const float*)d_in[1];
  const float* scales   = (const float*)d_in[2];
  const int*   indices  = (const int*)d_in[3];
  const int*   signs    = (const int*)d_in[4];
  float*       out      = (float*)d_out;

  __hip_bfloat16* xb = (__hip_bfloat16*)d_ws;
  __hip_bfloat16* wb = (__hip_bfloat16*)((char*)d_ws + (size_t)M_TOTAL * IN_DIM * 2);

  // 1) x -> bf16
  {
    const int total = M_TOTAL * IN_DIM;           // 33,554,432
    const int blocks = total / (256 * 8);         // 16384
    cvt_x_kernel<<<blocks, 256, 0, stream>>>(x, xb);
  }
  // 2) decode + inverse-Haar + scale -> bf16 weights
  decode_w_kernel<<<OUT_DIM, 512, 0, stream>>>(codebook, scales, indices, signs, wb);
  // 3) WMMA GEMM (double-buffered async-LDS pipeline)
  {
    dim3 grid(OUT_DIM / NT, M_TOTAL / MT);        // 32 x 64
    gemm_bf16_kernel<<<grid, 256, 0, stream>>>(xb, wb, out);
  }
}
